// MHAttention_15401752724219
// MI455X (gfx1250) — compile-verified, matrix operand order checked by reading, not run on hardware
//
#include <hip/hip_runtime.h>

typedef __attribute__((ext_vector_type(16))) __bf16 v16bf;
typedef __attribute__((ext_vector_type(8)))  float  v8f;
typedef __attribute__((ext_vector_type(4)))  unsigned int v4u;
typedef __attribute__((ext_vector_type(8)))  int v8i;
typedef __attribute__((ext_vector_type(4)))  int v4i;

#define B_  4
#define S_  4096
#define C_  1024
#define H_  16
#define D_  64
#define DK_ 256

#if __has_builtin(__builtin_amdgcn_tensor_load_to_lds) && __has_builtin(__builtin_amdgcn_s_wait_tensorcnt)
#define HAVE_TDM 1
#else
#define HAVE_TDM 0
#endif

__device__ __forceinline__ unsigned short f2bf(float f) {
  unsigned int u = __float_as_uint(f);
  u += 0x7FFFu + ((u >> 16) & 1u);
  return (unsigned short)(u >> 16);
}

__global__ void cvt_f32_bf16(const float* __restrict__ in,
                             unsigned short* __restrict__ out, size_t n) {
  size_t i = (size_t)blockIdx.x * blockDim.x + threadIdx.x;
  size_t stride = (size_t)gridDim.x * blockDim.x;
  for (; i < n; i += stride) out[i] = f2bf(in[i]);
}

// ---------------------------------------------------------------------------
// Generic bf16 GEMM: Y[m,n] = sum_k A[m,k] * (TRANS_B ? B[n,k] : B[k,n])
// 256 threads = 8 waves; each wave computes a 32x64 tile (2x4 wmma frags).
// Double-buffered LDS pipeline: global loads for tile k+1 are in flight
// while WMMAs consume tile k from LDS. Batched via blockIdx.z.
// ---------------------------------------------------------------------------
template<int BM, int BN, bool TRANS_B, bool OUT_BF16>
__global__ __launch_bounds__(256)
void gemm_bf16(const unsigned short* __restrict__ A,
               const unsigned short* __restrict__ Bm,
               void* __restrict__ Y,
               int K, int lda, int ldb, int ldy,
               int batchH,
               size_t aSB, size_t aSH, size_t bSB, size_t bSH,
               size_t ySB, size_t ySH,
               const float* __restrict__ bias_n,
               const float* __restrict__ bias_m) {
  constexpr int WN = BN / 64;       // waves along N
  constexpr int WM = 8 / WN;        // waves along M
  static_assert(WM * 32 == BM, "tile mismatch");
  constexpr int ACH = BM * 4 / 256; // uint4 chunks of A per thread
  constexpr int BCH = BN * 4 / 256; // uint4 chunks of B per thread

  __shared__ alignas(16) unsigned short As[2][BM][32];
  __shared__ alignas(16) unsigned short Bt[2][BN][32];   // n-major

  const int tid  = threadIdx.x;
  const int w    = tid >> 5;
  const int lane = tid & 31;
  const int half = (lane < 16) ? 0 : 16;
  const int l16  = lane & 15;
  const int wm   = w / WN;
  const int wn   = w % WN;

  const int zb = blockIdx.z / batchH;
  const int zh = blockIdx.z % batchH;
  const unsigned short* gA = A  + zb * aSB + zh * aSH + (size_t)blockIdx.y * BM * lda;
  const unsigned short* gB = Bm + zb * bSB + zh * bSH;
  const size_t yOff = zb * ySB + zh * ySH;

  uint4 ar[ACH], br[BCH];

  auto fetch = [&](int k0) {
#pragma unroll
    for (int i = 0; i < ACH; ++i) {
      int c = tid + i * 256;
      int row = c >> 2, col8 = (c & 3) * 8;
      const unsigned short* p = gA + (size_t)row * lda + k0 + col8;
      ar[i] = *(const uint4*)p;
      __builtin_prefetch(p + 64, 0, 1);   // two k-tiles ahead
    }
#pragma unroll
    for (int i = 0; i < BCH; ++i) {
      int c = tid + i * 256;
      if constexpr (TRANS_B) {
        int row = c >> 2, col8 = (c & 3) * 8;
        const unsigned short* p = gB + ((size_t)blockIdx.x * BN + row) * ldb + k0 + col8;
        br[i] = *(const uint4*)p;
        __builtin_prefetch(p + 64, 0, 1);
      } else {
        int kk = c / (BN / 8), n8 = (c % (BN / 8)) * 8;
        br[i] = *(const uint4*)(gB + (size_t)(k0 + kk) * ldb + blockIdx.x * BN + n8);
      }
    }
  };
  auto commit = [&](int buf) {
#pragma unroll
    for (int i = 0; i < ACH; ++i) {
      int c = tid + i * 256;
      int row = c >> 2, col8 = (c & 3) * 8;
      *(uint4*)&As[buf][row][col8] = ar[i];
    }
#pragma unroll
    for (int i = 0; i < BCH; ++i) {
      int c = tid + i * 256;
      if constexpr (TRANS_B) {
        int row = c >> 2, col8 = (c & 3) * 8;
        *(uint4*)&Bt[buf][row][col8] = br[i];
      } else {
        int kk = c / (BN / 8), n8 = (c % (BN / 8)) * 8;
        union { uint4 v; unsigned short s[8]; } u; u.v = br[i];
#pragma unroll
        for (int j = 0; j < 8; ++j) Bt[buf][n8 + j][kk] = u.s[j];
      }
    }
  };

  const v8f vzero = {};
  v8f acc[2][4];
  for (int i = 0; i < 2; ++i)
    for (int j = 0; j < 4; ++j) acc[i][j] = vzero;

  fetch(0);
  commit(0);
  int buf = 0;
  for (int k0 = 0; k0 < K; k0 += 32) {
    const bool hasNext = (k0 + 32 < K);
    if (hasNext) fetch(k0 + 32);       // global loads in flight during compute
    __syncthreads();

    v16bf af[2], bfv[4];
#pragma unroll
    for (int i = 0; i < 2; ++i)
      af[i] = *(const v16bf*)&As[buf][wm * 32 + i * 16 + l16][half];
#pragma unroll
    for (int j = 0; j < 4; ++j)
      bfv[j] = *(const v16bf*)&Bt[buf][wn * 64 + j * 16 + l16][half];
#pragma unroll
    for (int i = 0; i < 2; ++i)
#pragma unroll
      for (int j = 0; j < 4; ++j)
        acc[i][j] = __builtin_amdgcn_wmma_f32_16x16x32_bf16(
            false, af[i], false, bfv[j], (short)0, acc[i][j], false, false);

    if (hasNext) commit(buf ^ 1);
    buf ^= 1;
  }

  float* Yf = (float*)Y;
  unsigned short* Yb = (unsigned short*)Y;
#pragma unroll
  for (int i = 0; i < 2; ++i) {
#pragma unroll
    for (int j = 0; j < 4; ++j) {
#pragma unroll
      for (int r = 0; r < 8; ++r) {
        int gm = blockIdx.y * BM + wm * 32 + i * 16 + ((lane < 16) ? r : 8 + r);
        int gn = blockIdx.x * BN + wn * 64 + j * 16 + l16;
        float v = acc[i][j][r];
        if (bias_n) v += bias_n[gn];
        if (bias_m) v += bias_m[gm];
        size_t idx = yOff + (size_t)gm * ldy + gn;
        if constexpr (OUT_BF16) Yb[idx] = f2bf(v); else Yf[idx] = v;
      }
    }
  }
}

// ---------------------------------------------------------------------------
// Fused Linformer attention per (b,h) and 64-row s-tile:
//   scores = Q Kp^T / 8 ; P = softmax_k(scores) ; head = P Vp
// Kp staged into LDS via the Tensor Data Mover (D# built per ISA ch.8);
// Vp staged d-major (transposed) by hand. 128 threads = 4 waves.
// ---------------------------------------------------------------------------
__global__ __launch_bounds__(128)
void linformer_attn(const unsigned short* __restrict__ qkv,
                    const unsigned short* __restrict__ Kp,
                    const unsigned short* __restrict__ Vp,
                    unsigned short* __restrict__ head) {
  __shared__ alignas(16) unsigned short Kp_s[DK_][D_];   // 32 KB, reused for P
  __shared__ alignas(16) unsigned short Vpt_s[D_][DK_];  // 32 KB, d-major

  const int tid  = threadIdx.x;
  const int w    = tid >> 5;
  const int lane = tid & 31;
  const int half = (lane < 16) ? 0 : 16;
  const int l16  = lane & 15;
  const int bh   = blockIdx.y;
  const int b    = bh >> 4;    // / H_
  const int h    = bh & 15;    // % H_
  const int sBlock = blockIdx.x * 64;

  const unsigned short* gKp = Kp + (size_t)bh * DK_ * D_;
  const unsigned short* gVp = Vp + (size_t)bh * DK_ * D_;
  unsigned short* KpFlat = &Kp_s[0][0];

#if HAVE_TDM
  if (w == 0) {
    // Tensor DMA: contiguous 256x64 bf16 tile, global -> LDS.
    unsigned long long ga = (unsigned long long)gKp;
    unsigned int ldsOff = (unsigned int)(unsigned long long)KpFlat; // LDS aperture: addr[31:0]
    v4u g0;
    g0[0] = 1u;                                   // count=1, user desc
    g0[1] = ldsOff;                               // lds_addr
    g0[2] = (unsigned int)ga;                     // global_addr[31:0]
    g0[3] = (unsigned int)((ga >> 32) & 0x1FFFFFFu) | 0x80000000u; // ga[56:32], type=2
    v8i g1;
    g1[0] = 1 << 16;                              // data_size = 2 bytes
    g1[1] = (D_ & 0xFFFF) << 16;                  // tensor_dim0[15:0]
    g1[2] = (DK_ & 0xFFFF) << 16;                 // tensor_dim0[31:16]=0 | tensor_dim1[15:0]
    g1[3] = (D_ << 16);                           // tensor_dim1[31:16]=0 | tile_dim0=64
    g1[4] = DK_;                                  // tile_dim1=256, tile_dim2=0
    g1[5] = D_;                                   // tensor_dim0_stride[31:0]=64
    g1[6] = 0;                                    // stride hi / dim1_stride lo
    g1[7] = 0;
    v4i gz = {0, 0, 0, 0};
#if defined(__clang_major__) && __clang_major__ >= 23
    v8i gz8 = {0, 0, 0, 0, 0, 0, 0, 0};
    __builtin_amdgcn_tensor_load_to_lds(g0, g1, gz, gz, gz8, 0);
#else
    __builtin_amdgcn_tensor_load_to_lds(g0, g1, gz, gz, 0);
#endif
    __builtin_amdgcn_s_wait_tensorcnt(0);
  }
#else
  for (int c = tid; c < DK_ * D_ / 8; c += 128)
    *(uint4*)(KpFlat + c * 8) = *(const uint4*)(gKp + c * 8);
#endif
  for (int c = tid; c < DK_ * D_ / 8; c += 128) {
    int k = c >> 3, d0 = (c & 7) * 8;
    union { uint4 v; unsigned short s[8]; } u;
    u.v = *(const uint4*)(gVp + (size_t)k * D_ + d0);
#pragma unroll
    for (int j = 0; j < 8; ++j) Vpt_s[d0 + j][k] = u.s[j];
  }
  __syncthreads();

  // Q fragments (A layout): lane = row, k-half by lane group; d = 0..63
  const size_t qrow = ((size_t)(b * S_ + sBlock + w * 16 + l16)) * (3 * H_ * D_) + h * D_;
  v16bf a0 = *(const v16bf*)(qkv + qrow + half);
  v16bf a1 = *(const v16bf*)(qkv + qrow + 32 + half);

  const v8f vzero = {};
  v8f sc[16];
#pragma unroll
  for (int kt = 0; kt < 16; ++kt) sc[kt] = vzero;
#pragma unroll
  for (int kt = 0; kt < 16; ++kt) {
    const int kRow = kt * 16 + l16;
    v16bf b0 = *(const v16bf*)&Kp_s[kRow][half];        // B[d,k]=Kp[k][d]
    v16bf b1 = *(const v16bf*)&Kp_s[kRow][32 + half];
    sc[kt] = __builtin_amdgcn_wmma_f32_16x16x32_bf16(false, a0, false, b0, (short)0, sc[kt], false, false);
    sc[kt] = __builtin_amdgcn_wmma_f32_16x16x32_bf16(false, a1, false, b1, (short)0, sc[kt], false, false);
  }

  // softmax over k (scale 1/sqrt(64) folded into exp2 factor)
  const float c0 = 0.18033688011112042f;  // log2(e)/8
  float rsum[8];
#pragma unroll
  for (int i = 0; i < 8; ++i) {
    float mx = -3.0e38f;
#pragma unroll
    for (int kt = 0; kt < 16; ++kt) mx = fmaxf(mx, sc[kt][i]);
#pragma unroll
    for (int msk = 1; msk < 16; msk <<= 1) mx = fmaxf(mx, __shfl_xor(mx, msk, 32));
    float s = 0.f;
#pragma unroll
    for (int kt = 0; kt < 16; ++kt) {
      float p = exp2f((sc[kt][i] - mx) * c0);
      sc[kt][i] = p;
      s += p;
    }
#pragma unroll
    for (int msk = 1; msk < 16; msk <<= 1) s += __shfl_xor(s, msk, 32);
    rsum[i] = s;
  }

  __syncthreads();  // all waves finished reading Kp_s; reuse it for P

  unsigned short* P = KpFlat + w * (16 * DK_);  // this wave's 16x256 P strip
#pragma unroll
  for (int kt = 0; kt < 16; ++kt)
#pragma unroll
    for (int i = 0; i < 8; ++i) {
      int r = (lane < 16) ? i : 8 + i;
      P[r * DK_ + kt * 16 + l16] = f2bf(sc[kt][i]);
    }

  v8f ho[4];
#pragma unroll
  for (int nt = 0; nt < 4; ++nt) ho[nt] = vzero;
#pragma unroll
  for (int kk = 0; kk < 8; ++kk) {
    v16bf ap = *(const v16bf*)(P + l16 * DK_ + kk * 32 + half);
#pragma unroll
    for (int nt = 0; nt < 4; ++nt) {
      v16bf bp = *(const v16bf*)&Vpt_s[nt * 16 + l16][kk * 32 + half];  // B[k,d]=Vpt[d][k]
      ho[nt] = __builtin_amdgcn_wmma_f32_16x16x32_bf16(false, ap, false, bp, (short)0, ho[nt], false, false);
    }
  }

#pragma unroll
  for (int nt = 0; nt < 4; ++nt)
#pragma unroll
    for (int i = 0; i < 8; ++i) {
      int m = (lane < 16) ? i : 8 + i;
      int srow = sBlock + w * 16 + m;
      float v = ho[nt][i] / rsum[i];
      head[(((size_t)(b * S_ + srow)) * H_ + h) * D_ + nt * 16 + l16] = f2bf(v);
    }
}

// ---------------------------------------------------------------------------
extern "C" void kernel_launch(void* const* d_in, const int* in_sizes, int n_in,
                              void* d_out, int out_size, void* d_ws, size_t ws_size,
                              hipStream_t stream) {
  const float* x   = (const float*)d_in[0];
  const float* Wq  = (const float*)d_in[1];
  const float* Wk  = (const float*)d_in[2];
  const float* Wv  = (const float*)d_in[3];
  const float* E_w = (const float*)d_in[4];
  const float* E_b = (const float*)d_in[5];
  const float* F_w = (const float*)d_in[6];
  const float* F_b = (const float*)d_in[7];
  const float* Wo  = (const float*)d_in[8];
  const float* bo  = (const float*)d_in[9];
  float* out = (float*)d_out;

  char* ws = (char*)d_ws;
  size_t off = 0;
  auto alloc = [&](size_t elems) {
    unsigned short* p = (unsigned short*)(ws + off);
    off += ((elems * 2 + 255) / 256) * 256;
    return p;
  };
  unsigned short* xb   = alloc((size_t)B_ * S_ * C_);
  unsigned short* Wqkv = alloc((size_t)3 * H_ * D_ * C_);
  unsigned short* Ewb  = alloc((size_t)DK_ * S_);
  unsigned short* Fwb  = alloc((size_t)DK_ * S_);
  unsigned short* Wob  = alloc((size_t)C_ * H_ * D_);
  unsigned short* QKV  = alloc((size_t)B_ * S_ * 3 * H_ * D_);   // [B,S, 3,H,D]
  unsigned short* Kp   = alloc((size_t)B_ * H_ * DK_ * D_);
  unsigned short* Vp   = alloc((size_t)B_ * H_ * DK_ * D_);
  unsigned short* head = alloc((size_t)B_ * S_ * H_ * D_);       // [B,S,H,D]

  const size_t HDC = (size_t)H_ * D_ * C_;
  cvt_f32_bf16<<<2048, 256, 0, stream>>>(x, xb, (size_t)B_ * S_ * C_);
  cvt_f32_bf16<<<512, 256, 0, stream>>>(Wq, Wqkv, HDC);
  cvt_f32_bf16<<<512, 256, 0, stream>>>(Wk, Wqkv + HDC, HDC);
  cvt_f32_bf16<<<512, 256, 0, stream>>>(Wv, Wqkv + 2 * HDC, HDC);
  cvt_f32_bf16<<<512, 256, 0, stream>>>(E_w, Ewb, (size_t)DK_ * S_);
  cvt_f32_bf16<<<512, 256, 0, stream>>>(F_w, Fwb, (size_t)DK_ * S_);
  cvt_f32_bf16<<<512, 256, 0, stream>>>(Wo, Wob, (size_t)C_ * H_ * D_);

  // 1) fused QKV projection: [B*S, C] x [3HD, C]^T -> bf16 [B*S, 3HD]
  {
    dim3 g(3 * H_ * D_ / 128, B_ * S_ / 128, 1);
    gemm_bf16<128, 128, true, true><<<g, 256, 0, stream>>>(
        xb, Wqkv, QKV, C_, C_, C_, 3 * H_ * D_, 1,
        0, 0, 0, 0, 0, 0, nullptr, nullptr);
  }
  // 2) Kp[bh][k,d] = sum_s E_w[k,s] * K[bh][s,d]  (batched, strided B)
  {
    dim3 g(1, 1, B_ * H_);
    gemm_bf16<256, 64, false, true><<<g, 256, 0, stream>>>(
        Ewb, QKV + H_ * D_, Kp, S_, S_, 3 * H_ * D_, D_, H_,
        0, 0, (size_t)S_ * 3 * H_ * D_, (size_t)D_,
        (size_t)H_ * DK_ * D_, (size_t)DK_ * D_, nullptr, E_b);
    gemm_bf16<256, 64, false, true><<<g, 256, 0, stream>>>(
        Fwb, QKV + 2 * H_ * D_, Vp, S_, S_, 3 * H_ * D_, D_, H_,
        0, 0, (size_t)S_ * 3 * H_ * D_, (size_t)D_,
        (size_t)H_ * DK_ * D_, (size_t)DK_ * D_, nullptr, F_b);
  }
  // 3) fused scores -> softmax -> P*Vp
  {
    dim3 g(S_ / 64, B_ * H_, 1);
    linformer_attn<<<g, 128, 0, stream>>>(QKV, Kp, Vp, head);
  }
  // 4) output projection: [B*S, HD] x [C, HD]^T + bo -> f32 d_out
  {
    dim3 g(C_ / 128, B_ * S_ / 128, 1);
    gemm_bf16<128, 128, true, false><<<g, 256, 0, stream>>>(
        head, Wob, out, H_ * D_, H_ * D_, H_ * D_, C_, 1,
        0, 0, 0, 0, 0, 0, bo, nullptr);
  }
}
